// KERNContext_35862976921811
// MI455X (gfx1250) — matrix operand order repaired and checked by reading, not compiled, not run on hardware
//
#include <hip/hip_runtime.h>
#include <hip/hip_bf16.h>

// ---------------------------------------------------------------------------
// GGNN-obj on MI455X (gfx1250): all matmuls via v_wmma_f32_16x16x32_bf16.
// All tile edges padded so the GEMM inner loop has zero guards/fallbacks.
// ---------------------------------------------------------------------------

typedef __bf16 bf16;
typedef __attribute__((ext_vector_type(16))) __bf16 v16bf;
typedef __attribute__((ext_vector_type(8)))  __bf16 v8bf;
typedef __attribute__((ext_vector_type(8)))  float  v8f;

#define NOBJ   256
#define NCLS   151
#define HDIM   512
#define TSTEP  3
#define NCROWS (NOBJ * NCLS)        // 38656 = 302 * 128
#define CLSK   (NCLS * HDIM)        // 77312
#define KPAD   160                  // padded message-passing K (151 -> 160)
#define MPAD   384                  // padded Mcat rows (302 -> 3*128)
#define CLSNP  256                  // padded classifier N (151 -> 2*128)
#define KSPL   16                   // classifier split-K factor
#define KCHNK  (CLSK / KSPL)        // 4832, multiple of 32

#define BM 128
#define BN 128
#define BK 32

__device__ __forceinline__ float fsigmoid(float x) {
  x = fminf(fmaxf(x, -30.f), 30.f);
  return 1.f / (1.f + __expf(-x));
}
__device__ __forceinline__ float ftanhf(float x) {
  x = fminf(fmaxf(x, -15.f), 15.f);
  float e = __expf(2.f * x);
  return (e - 1.f) / (e + 1.f);
}

// ---------------------------------------------------------------------------
// Tiled WMMA GEMM.  C[M,N](+epilogue) = A[M,K] * B
// A = concat(A0, A1) along K (switch at K0); all loads unconditional (padded).
// BT=0: B is [K][N] row-major (transposed into LDS)  -- message GEMM
// BT=1: B is [N][K] row-major (direct LDS store)      -- weight GEMMs
// MODE: 0 = AV (batched z, interleaved av1|av2 bf16)
//       1 = ZR gate (sigmoid -> zv bf16, rv*h bf16)
//       2 = HV gate (tanh -> GRU update, hidden f32+bf16)
//       3 = OUT  (relu -> bf16; A1 row = row/NCLS)
//       4 = CLS  (split-K over z, f32 partials)
// ---------------------------------------------------------------------------
struct GP {
  const bf16* A0; const bf16* A1;
  int lda0, lda1, K0;
  const bf16* B; int ldb; long bz;     // MODE 0: B += z*bz
  int kchunk;                          // MODE 4: K-range per z
  int M, N, K;
  const float* bias;
  bf16*  av;
  bf16*  zv;
  bf16*  rh;
  const float* hidf_r;
  float* hidf_w;
  bf16*  hidb_w;
  bf16*  outb;
  float* outf;
};

template<int MODE, int BT>
__global__ __launch_bounds__(256)
void gemm_wmma(GP p) {
  __shared__ bf16 sA [BM][BK + 8];     // 80B rows -> 16B-aligned frag loads
  __shared__ bf16 sBt[BN][BK + 8];     // [n][k]

  const int t    = threadIdx.x;
  const int lane = t & 31;
  const int w    = t >> 5;             // 8 waves
  const int wm   = w >> 1;             // 0..3 -> 32 rows each
  const int wn   = w & 1;              // 0..1 -> 64 cols each
  const int bm0  = blockIdx.y * BM;
  const int bn0  = blockIdx.x * BN;
  const int nz   = blockIdx.z;
  const bf16* B  = (MODE == 0) ? p.B + (long)nz * p.bz : p.B;
  const int kbeg = (MODE == 4) ? nz * p.kchunk : 0;
  const int kend = (MODE == 4) ? kbeg + p.kchunk : p.K;

  v8f acc[2][4];
#pragma unroll
  for (int i = 0; i < 2; ++i)
#pragma unroll
    for (int j = 0; j < 4; ++j)
#pragma unroll
      for (int e = 0; e < 8; ++e) acc[i][j][e] = 0.f;

  v8bf ra[2], rb[2];

  auto loadA = [&](int k0) {
#pragma unroll
    for (int pas = 0; pas < 2; ++pas) {
      const int grow = bm0 + (t >> 2) + pas * 64;
      const int gk   = k0 + (t & 3) * 8;
      const bf16* src = (gk < p.K0)
          ? p.A0 + (long)grow * p.lda0 + gk
          : p.A1 + ((MODE == 3) ? (long)(grow / NCLS) : (long)grow) * p.lda1
                 + (gk - p.K0);
      ra[pas] = *(const v8bf*)src;
    }
  };
  auto loadB = [&](int k0) {
    if (BT == 0) {
#pragma unroll
      for (int pas = 0; pas < 2; ++pas) {
        const int kr = (t >> 4) + pas * 16;
        const int cg = (t & 15) * 8;
        rb[pas] = *(const v8bf*)(B + (long)(k0 + kr) * p.ldb + (bn0 + cg));
      }
    } else {
#pragma unroll
      for (int pas = 0; pas < 2; ++pas) {
        const int j  = (t >> 2) + pas * 64;
        const int kc = (t & 3) * 8;
        rb[pas] = *(const v8bf*)(B + (long)(bn0 + j) * p.ldb + k0 + kc);
      }
    }
  };
  auto storeTiles = [&]() {
#pragma unroll
    for (int pas = 0; pas < 2; ++pas)
      *(v8bf*)&sA[(t >> 2) + pas * 64][(t & 3) * 8] = ra[pas];
    if (BT == 0) {
#pragma unroll
      for (int pas = 0; pas < 2; ++pas) {
        const int kr = (t >> 4) + pas * 16;
        const int cg = (t & 15) * 8;
#pragma unroll
        for (int jj = 0; jj < 8; ++jj) sBt[cg + jj][kr] = rb[pas][jj];
      }
    } else {
#pragma unroll
      for (int pas = 0; pas < 2; ++pas)
        *(v8bf*)&sBt[(t >> 2) + pas * 64][(t & 3) * 8] = rb[pas];
    }
  };

  union Frag { v16bf v; v8bf h[2]; };

  loadA(kbeg);
  loadB(kbeg);
  for (int k0 = kbeg; k0 < kend; k0 += BK) {
    storeTiles();
    __syncthreads();
    const int kn = k0 + BK;
    if (kn < kend) { loadA(kn); loadB(kn); }   // overlap VMEM with WMMA

    Frag af[2];
#pragma unroll
    for (int ti = 0; ti < 2; ++ti) {
      const int ar = wm * 32 + ti * 16 + (lane & 15);
      const int ko = (lane >> 4) * 8;          // lanes 0-15: K0-7/16-23
      af[ti].h[0] = *(const v8bf*)&sA[ar][ko];
      af[ti].h[1] = *(const v8bf*)&sA[ar][ko + 16];
    }
#pragma unroll
    for (int tj = 0; tj < 4; ++tj) {
      Frag bf;
      const int bc = wn * 64 + tj * 16 + (lane & 15);
      const int kb = (lane >> 4) * 8;          // lanes 0-15: K0-7/16-23
      bf.h[0] = *(const v8bf*)&sBt[bc][kb];
      bf.h[1] = *(const v8bf*)&sBt[bc][kb + 16];
#pragma unroll
      for (int ti = 0; ti < 2; ++ti)
        acc[ti][tj] = __builtin_amdgcn_wmma_f32_16x16x32_bf16(
            false, af[ti].v, false, bf.v, (short)0, acc[ti][tj], false, false);
    }
    __syncthreads();
  }

  // ---- epilogue ----------------------------------------------------------
#pragma unroll
  for (int ti = 0; ti < 2; ++ti)
#pragma unroll
    for (int tj = 0; tj < 4; ++tj)
#pragma unroll
      for (int i = 0; i < 8; ++i) {
        const int row = bm0 + wm * 32 + ti * 16 + ((lane >> 4) * 8) + i;
        const int col = bn0 + wn * 64 + tj * 16 + (lane & 15);
        if (row >= p.M || col >= p.N) continue;
        const float v = acc[ti][tj][i];
        if (MODE == 0) {
          const int  r2   = (row < NCLS) ? row : row - NCLS;
          const long base = ((long)nz * NCLS + r2) * (2 * HDIM)
                          + ((row < NCLS) ? 0 : HDIM);
          p.av[base + col] = (bf16)v;
        } else if (MODE == 1) {
          const float s = fsigmoid(v + p.bias[col]);
          if (col < HDIM) {
            p.zv[(long)row * HDIM + col] = (bf16)s;
          } else {
            const long idx = (long)row * HDIM + (col - HDIM);
            p.rh[idx] = (bf16)(s * p.hidf_r[idx]);
          }
        } else if (MODE == 2) {
          const float hv  = ftanhf(v + p.bias[col]);
          const long  idx = (long)row * HDIM + col;
          const float z   = (float)p.zv[idx];
          const float hn  = (1.f - z) * p.hidf_r[idx] + z * hv;
          p.hidf_w[idx] = hn;
          p.hidb_w[idx] = (bf16)hn;
        } else if (MODE == 3) {
          float o = v + p.bias[col];
          o = o > 0.f ? o : 0.f;
          p.outb[(long)row * HDIM + col] = (bf16)o;
        } else {
          p.outf[((long)nz * NOBJ + row) * NCLS + col] = v;  // split-K partial
        }
      }
}

// ---------------------------------------------------------------------------
// Packing / elementwise helper kernels (weights stay in natural [N][K] layout)
// ---------------------------------------------------------------------------
__global__ void k_pack_wzr(const float* w3w, const float* w3u,
                           const float* w4w, const float* w4u, bf16* W) {
  int id = blockIdx.x * 256 + threadIdx.x;       // [1024 j][1536 k]
  if (id >= 1024 * 1536) return;
  int j = id / 1536, k = id % 1536;
  float v;
  if (j < 512) v = (k < 1024) ? w3w[j * 1024 + k] : w3u[j * 512 + (k - 1024)];
  else { int jj = j - 512;
         v = (k < 1024) ? w4w[jj * 1024 + k] : w4u[jj * 512 + (k - 1024)]; }
  W[id] = (bf16)v;
}
__global__ void k_pack_whv(const float* w5w, const float* w5u, bf16* W) {
  int id = blockIdx.x * 256 + threadIdx.x;       // [512 j][1536 k]
  if (id >= 512 * 1536) return;
  int j = id / 1536, k = id % 1536;
  float v = (k < 1024) ? w5w[j * 1024 + k] : w5u[j * 512 + (k - 1024)];
  W[id] = (bf16)v;
}
__global__ void k_cvt_bf16(const float* src, bf16* dst, long n) {
  long id = (long)blockIdx.x * 256 + threadIdx.x;
  if (id < n) dst[id] = (bf16)src[id];
}
__global__ void k_pack_wcls(const float* wc, bf16* W) {
  long id = (long)blockIdx.x * 256 + threadIdx.x; // [256 j][77312 k]
  if (id >= (long)CLSNP * CLSK) return;
  long j = id / CLSK;
  W[id] = (j < NCLS) ? (bf16)wc[id] : (bf16)0.f;  // rows beyond 151 zero
}
__global__ void k_pack_mcat(const float* m, bf16* W) {
  int id = blockIdx.x * 256 + threadIdx.x;        // [384 r][160 c]
  if (id >= MPAD * KPAD) return;
  int r = id / KPAD, c = id % KPAD;
  float v = 0.f;
  if (c < NCLS && r < 2 * NCLS)
    v = (r < NCLS) ? m[c * NCLS + r] : m[(r - NCLS) * NCLS + c];
  W[id] = (bf16)v;
}
__global__ void k_pack_bias(const float* b3w, const float* b3u,
                            const float* b4w, const float* b4u,
                            const float* b5w, const float* b5u,
                            const float* bo,
                            float* bzr, float* bhv, float* bout) {
  int id = blockIdx.x * 256 + threadIdx.x;
  if (id < 512)        bzr[id] = b3w[id] + b3u[id];
  else if (id < 1024)  bzr[id] = b4w[id - 512] + b4u[id - 512];
  else if (id < 1536)  bhv[id - 1024] = b5w[id - 1024] + b5u[id - 1024];
  else if (id < 2048)  bout[id - 1536] = bo[id - 1536];
}
__global__ void k_init_hidden(const float* inp, float* hf, bf16* hb) {
  long id = (long)blockIdx.x * 256 + threadIdx.x;
  if (id >= (long)NCROWS * HDIM) return;
  long row = id / HDIM; int h = (int)(id % HDIM);
  int n = (int)(row / NCLS);
  float v = inp[(long)n * HDIM + h];
  hf[id] = v; hb[id] = (bf16)v;
}
__global__ void k_hidden_sum(const float* hf, float* hsum) {
  int id = blockIdx.x * 256 + threadIdx.x;
  if (id >= NCLS * HDIM) return;
  int c = id / HDIM, h = id % HDIM;
  float s = 0.f;
  for (int n = 0; n < NOBJ; ++n)
    s += hf[((long)n * NCLS + c) * HDIM + h];
  hsum[id] = s;
}
__global__ void k_make_x(const float* hf, const float* hsum, bf16* x) {
  long id = (long)blockIdx.x * 256 + threadIdx.x;  // [NOBJ][KPAD][HDIM]
  if (id >= (long)NOBJ * KPAD * HDIM) return;
  long rem = id % ((long)KPAD * HDIM);
  int n = (int)(id / ((long)KPAD * HDIM));
  int c = (int)(rem / HDIM), h = (int)(rem % HDIM);
  float v = 0.f;
  if (c < NCLS) v = hsum[c * HDIM + h] - hf[((long)n * NCLS + c) * HDIM + h];
  x[id] = (bf16)v;
}
__global__ void k_cls_reduce(const float* part, const float* bias, float* out) {
  int id = blockIdx.x * 256 + threadIdx.x;         // 256*151
  if (id >= NOBJ * NCLS) return;
  int row = id / NCLS, col = id % NCLS;
  float s = bias[col];
  for (int z = 0; z < KSPL; ++z)
    s += part[((long)z * NOBJ + row) * NCLS + col];
  out[id] = s;
}

// ---------------------------------------------------------------------------
extern "C" void kernel_launch(void* const* d_in, const int* in_sizes, int n_in,
                              void* d_out, int out_size, void* d_ws, size_t ws_size,
                              hipStream_t stream) {
  const float* input  = (const float*)d_in[0];
  const float* matrix = (const float*)d_in[1];
  const float* w3w = (const float*)d_in[2];  const float* b3w = (const float*)d_in[3];
  const float* w3u = (const float*)d_in[4];  const float* b3u = (const float*)d_in[5];
  const float* w4w = (const float*)d_in[6];  const float* b4w = (const float*)d_in[7];
  const float* w4u = (const float*)d_in[8];  const float* b4u = (const float*)d_in[9];
  const float* w5w = (const float*)d_in[10]; const float* b5w = (const float*)d_in[11];
  const float* w5u = (const float*)d_in[12]; const float* b5u = (const float*)d_in[13];
  const float* wout = (const float*)d_in[14]; const float* bout_i = (const float*)d_in[15];
  const float* wcls = (const float*)d_in[16]; const float* bcls_i = (const float*)d_in[17];
  float* out = (float*)d_out;

  char* ws = (char*)d_ws;
  size_t off = 0;
  auto carve = [&](size_t bytes) {
    char* q = ws + off;
    off = (off + bytes + 255) & ~(size_t)255;
    return q;
  };
  float* hid_f  = (float*)carve((size_t)NCROWS * HDIM * 4);
  bf16*  hid_b  = (bf16*) carve((size_t)NCROWS * HDIM * 2);
  bf16*  x_b    = (bf16*) carve((size_t)NOBJ * KPAD * HDIM * 2); // also `out` bf16
  bf16*  av_b   = (bf16*) carve((size_t)NCROWS * 2 * HDIM * 2);
  bf16*  zv_b   = (bf16*) carve((size_t)NCROWS * HDIM * 2);
  bf16*  rh_b   = (bf16*) carve((size_t)NCROWS * HDIM * 2);
  float* hsum   = (float*)carve((size_t)NCLS * HDIM * 4);
  bf16*  inp_b  = (bf16*) carve((size_t)NOBJ * HDIM * 2);
  bf16*  Wzr    = (bf16*) carve((size_t)1024 * 1536 * 2);
  bf16*  Whv    = (bf16*) carve((size_t)512 * 1536 * 2);
  bf16*  Wout   = (bf16*) carve((size_t)512 * 1024 * 2);
  bf16*  Wcls   = (bf16*) carve((size_t)CLSNP * CLSK * 2);
  bf16*  Mcat   = (bf16*) carve((size_t)MPAD * KPAD * 2);
  float* part   = (float*)carve((size_t)KSPL * NOBJ * NCLS * 4);
  float* bzr    = (float*)carve(1024 * 4);
  float* bhv    = (float*)carve(512 * 4);
  float* bout   = (float*)carve(512 * 4);
  (void)ws_size; (void)in_sizes; (void)n_in; (void)out_size;

  auto g1 = [](long n) { return dim3((unsigned)((n + 255) / 256)); };

  k_pack_wzr <<<g1(1024L*1536), 256, 0, stream>>>(w3w, w3u, w4w, w4u, Wzr);
  k_pack_whv <<<g1(512L*1536),  256, 0, stream>>>(w5w, w5u, Whv);
  k_cvt_bf16 <<<g1(512L*1024),  256, 0, stream>>>(wout, Wout, 512L*1024);
  k_pack_wcls<<<g1((long)CLSNP*CLSK), 256, 0, stream>>>(wcls, Wcls);
  k_pack_mcat<<<g1((long)MPAD*KPAD),  256, 0, stream>>>(matrix, Mcat);
  k_pack_bias<<<g1(2048),       256, 0, stream>>>(b3w, b3u, b4w, b4u, b5w, b5u,
                                                  bout_i, bzr, bhv, bout);
  k_init_hidden<<<g1((long)NCROWS*HDIM), 256, 0, stream>>>(input, hid_f, hid_b);
  k_cvt_bf16   <<<g1(NOBJ*HDIM),        256, 0, stream>>>(input, inp_b, NOBJ*HDIM);

  for (int t = 0; t < TSTEP; ++t) {
    k_hidden_sum<<<g1(NCLS*HDIM),             256, 0, stream>>>(hid_f, hsum);
    k_make_x    <<<g1((long)NOBJ*KPAD*HDIM),  256, 0, stream>>>(hid_f, hsum, x_b);

    {   // av[n] = Mcat(384x160, zero-padded) @ x[n](160x512)
      GP p{};
      p.A0 = Mcat; p.A1 = nullptr; p.lda0 = KPAD; p.lda1 = 0; p.K0 = 1 << 30;
      p.B = x_b; p.ldb = HDIM; p.bz = (long)KPAD * HDIM;
      p.M = 2 * NCLS; p.N = HDIM; p.K = KPAD;
      p.av = av_b;
      gemm_wmma<0, 0><<<dim3(HDIM / BN, MPAD / BM, NOBJ), 256, 0, stream>>>(p);
    }
    {   // [AV | h] @ Wzr^T -> sigmoid -> zv, rv*h
      GP p{};
      p.A0 = av_b; p.lda0 = 2 * HDIM; p.K0 = 2 * HDIM;
      p.A1 = hid_b; p.lda1 = HDIM;
      p.B = Wzr; p.ldb = 1536; p.bz = 0;
      p.M = NCROWS; p.N = 1024; p.K = 3 * HDIM;
      p.bias = bzr; p.zv = zv_b; p.rh = rh_b; p.hidf_r = hid_f;
      gemm_wmma<1, 1><<<dim3(1024 / BN, NCROWS / BM, 1), 256, 0, stream>>>(p);
    }
    {   // [AV | rv*h] @ Whv^T -> tanh -> GRU update
      GP p{};
      p.A0 = av_b; p.lda0 = 2 * HDIM; p.K0 = 2 * HDIM;
      p.A1 = rh_b; p.lda1 = HDIM;
      p.B = Whv; p.ldb = 1536; p.bz = 0;
      p.M = NCROWS; p.N = HDIM; p.K = 3 * HDIM;
      p.bias = bhv; p.zv = zv_b; p.hidf_r = hid_f;
      p.hidf_w = hid_f; p.hidb_w = hid_b;
      gemm_wmma<2, 1><<<dim3(HDIM / BN, NCROWS / BM, 1), 256, 0, stream>>>(p);
    }
  }

  {   // out = relu([hidden | inp_rep] @ Wout^T + b) -> bf16 (reuses x buffer)
    GP p{};
    p.A0 = hid_b; p.lda0 = HDIM; p.K0 = HDIM;
    p.A1 = inp_b; p.lda1 = HDIM;               // MODE 3: A1 row = row / NCLS
    p.B = Wout; p.ldb = 1024; p.bz = 0;
    p.M = NCROWS; p.N = HDIM; p.K = 2 * HDIM;
    p.bias = bout; p.outb = x_b;
    gemm_wmma<3, 1><<<dim3(HDIM / BN, NCROWS / BM, 1), 256, 0, stream>>>(p);
  }
  {   // obj_dists partials: split-K GEMM [256, 77312] x [77312, 151]
    GP p{};
    p.A0 = x_b; p.lda0 = CLSK; p.K0 = 1 << 30; p.A1 = nullptr;
    p.B = Wcls; p.ldb = CLSK; p.bz = 0; p.kchunk = KCHNK;
    p.M = NOBJ; p.N = NCLS; p.K = CLSK;
    p.outf = part;
    gemm_wmma<4, 1><<<dim3(CLSNP / BN, NOBJ / BM, KSPL), 256, 0, stream>>>(p);
  }
  k_cls_reduce<<<g1(NOBJ * NCLS), 256, 0, stream>>>(part, bcls_i, out);
}
